// DiffTransformerEncoder_90598040142451
// MI455X (gfx1250) — compile-verified, matrix operand order checked by reading, not run on hardware
//
#include <hip/hip_runtime.h>
#include <cstdint>
#include <cstddef>

typedef __attribute__((ext_vector_type(16))) _Float16 v16h;
typedef __attribute__((ext_vector_type(8)))  float    v8f;
typedef __attribute__((ext_vector_type(4)))  unsigned int u32x4;
typedef __attribute__((ext_vector_type(4)))  int      i32x4;
typedef __attribute__((ext_vector_type(8)))  int      i32x8;

// ---------------------------------------------------------------------------
// WMMA fragment element index (CDNA5 ISA 7.12.2, 16-bit 16x32 A / 32x16 B)
// lane L (<16): K in {0..7}U{16..23}; lane L+16: K in {8..15}U{24..31}
// ---------------------------------------------------------------------------
__device__ __forceinline__ int wmma_kidx(int vg, int s, int hiK) {
    return ((vg >> 2) << 4) + (hiK << 3) + ((vg & 3) << 1) + s;
}

__device__ __forceinline__ float gelu_f(float x) {
    float x3 = x * x * x;
    return 0.5f * x * (1.0f + tanhf(0.7978845608028654f * (x + 0.044715f * x3)));
}

// ---------------------------------------------------------------------------
// Tensor Data Mover: 1-D contiguous DMA of `nelem` 2-byte elements to LDS.
// D# per CDNA5 ISA: count=1, data_size=2B,
// tile_dim0 = tensor_dim0 = stride = nelem, 1-D tile, type=2 ("image").
// ---------------------------------------------------------------------------
__device__ __forceinline__ void tdm_load_1d(unsigned lds_off, const void* g, unsigned nelem) {
    unsigned long long ga = (unsigned long long)(size_t)g;
    u32x4 g0;
    g0[0] = 1u;                                              // count=1
    g0[1] = lds_off;                                         // lds_addr
    g0[2] = (unsigned)ga;                                    // global_addr[31:0]
    g0[3] = (unsigned)((ga >> 32) & 0x1ffffffu) | (2u << 30);// addr[56:32] | type=2
    i32x8 g1;
    g1[0] = 0x00010000;                                      // data_size=1 (2 bytes)
    g1[1] = (int)((nelem & 0xffffu) << 16);                  // tensor_dim0[15:0]
    g1[2] = (int)(((nelem >> 16) & 0xffffu) | (1u << 16));   // tdim0 hi | tensor_dim1=1
    g1[3] = (int)((nelem & 0xffffu) << 16);                  // tile_dim0
    g1[4] = 0;                                               // tile_dim1/2 unused
    g1[5] = (int)nelem;                                      // tensor_dim0_stride lo
    g1[6] = 0;
    g1[7] = 0;
    i32x4 z4 = {};
#if defined(__clang_major__) && __clang_major__ >= 23
    i32x8 z8 = {};
    __builtin_amdgcn_tensor_load_to_lds(g0, g1, z4, z4, z8, 0);
#else
    __builtin_amdgcn_tensor_load_to_lds(g0, g1, z4, z4, 0);
#endif
}

// ---------------------------------------------------------------------------
// Patch-embed conv (direct). Output in (B, H*W, C) token layout.
// ---------------------------------------------------------------------------
__global__ void conv_kernel(const float* __restrict__ in, const float* __restrict__ w,
                            const float* __restrict__ bias, float* __restrict__ out,
                            int Bn, int Cin, int Hin, int Win, int Cout,
                            int Ho, int Wo, int ks, int st, int pd, int nchw) {
    int idx = blockIdx.x * blockDim.x + threadIdx.x;
    int total = Bn * Ho * Wo * Cout;
    if (idx >= total) return;
    int c  = idx % Cout;
    int xx = (idx / Cout) % Wo;
    int yy = (idx / (Cout * Wo)) % Ho;
    int b  = idx / (Cout * Wo * Ho);
    float acc = bias[c];
    for (int ci = 0; ci < Cin; ++ci)
        for (int ky = 0; ky < ks; ++ky) {
            int iy = yy * st - pd + ky;
            if (iy < 0 || iy >= Hin) continue;
            for (int kx = 0; kx < ks; ++kx) {
                int ix = xx * st - pd + kx;
                if (ix < 0 || ix >= Win) continue;
                float iv = nchw ? in[(((size_t)b * Cin + ci) * Hin + iy) * Win + ix]
                                : in[(((size_t)b * Hin + iy) * Win + ix) * Cin + ci];
                acc += iv * w[(((size_t)c * Cin + ci) * ks + ky) * ks + kx];
            }
        }
    out[idx] = acc;
}

// ---------------------------------------------------------------------------
// Fused LayerNorm + RMSNorm, in place on (M, C). One wave per row.
// ---------------------------------------------------------------------------
__global__ __launch_bounds__(256) void ln_rms_kernel(float* __restrict__ t,
                                                     const float* __restrict__ w,
                                                     const float* __restrict__ b,
                                                     int M, int C) {
    int wid = threadIdx.x >> 5, lane = threadIdx.x & 31;
    int row = blockIdx.x * 8 + wid;
    if (row >= M) return;
    float* x = t + (size_t)row * C;
    int nper = C >> 5;
    float loc[8];
    float s = 0.f, sq = 0.f;
    for (int i = 0; i < nper; ++i) {
        float v = x[lane + i * 32];
        loc[i] = v; s += v; sq += v * v;
    }
    for (int o = 16; o > 0; o >>= 1) { s += __shfl_xor(s, o); sq += __shfl_xor(sq, o); }
    float mu = s / C;
    float var = sq / C - mu * mu;
    float rs = rsqrtf(var + 1e-5f);
    float nrm = 0.f;
    for (int i = 0; i < nper; ++i) {
        int c = lane + i * 32;
        float lv = (loc[i] - mu) * rs * w[c] + b[c];
        loc[i] = lv; nrm += lv * lv;
    }
    for (int o = 16; o > 0; o >>= 1) nrm += __shfl_xor(nrm, o);
    float inv = sqrtf((float)C) / fmaxf(sqrtf(nrm), 1e-12f);
    for (int i = 0; i < nper; ++i) x[lane + i * 32] = loc[i] * inv;
}

// RMSNorm (M,C) fp32 -> f16 linear
__global__ __launch_bounds__(256) void rms_half_kernel(const float* __restrict__ t,
                                                       _Float16* __restrict__ out,
                                                       int M, int C) {
    int wid = threadIdx.x >> 5, lane = threadIdx.x & 31;
    int row = blockIdx.x * 8 + wid;
    if (row >= M) return;
    const float* x = t + (size_t)row * C;
    int nper = C >> 5;
    float loc[8];
    float sq = 0.f;
    for (int i = 0; i < nper; ++i) { float v = x[lane + i * 32]; loc[i] = v; sq += v * v; }
    for (int o = 16; o > 0; o >>= 1) sq += __shfl_xor(sq, o);
    float inv = sqrtf((float)C) / fmaxf(sqrtf(sq), 1e-12f);
    for (int i = 0; i < nper; ++i)
        out[(size_t)row * C + lane + i * 32] = (_Float16)(loc[i] * inv);
}

// ---------------------------------------------------------------------------
// Fragment swizzlers: lay data out so one WMMA operand = contiguous 32B/lane.
// A-frag buffer:  [(row_tile*chunks + chunk)*512 + lane*16 + e]
// B-frag buffer:  [(chunk*ntiles  + ntile)*512 + lane*16 + e]
// ---------------------------------------------------------------------------
__global__ void swizzleA_kernel(const _Float16* __restrict__ in, _Float16* __restrict__ out,
                                int M, int K) {
    int idx = blockIdx.x * blockDim.x + threadIdx.x;
    int total = M * K;
    if (idx >= total) return;
    int e    = idx & 15;
    int lane = (idx >> 4) & 31;
    int frag = idx >> 9;
    int chunks = K >> 5;
    int chunk = frag % chunks;
    int rt    = frag / chunks;
    int hiK = lane >> 4, n16 = lane & 15;
    int kf = wmma_kidx(e >> 1, e & 1, hiK);
    out[idx] = in[(size_t)(rt * 16 + n16) * K + chunk * 32 + kf];
}

// QKV weight (h, C, 2d) fp32 -> B-frag f16 over (K=C, Ncol=h*2d)
__global__ void pack_qkv_sw_kernel(const float* __restrict__ w, _Float16* __restrict__ out,
                                   int h, int C, int d2) {
    int Ncol = h * d2;
    int idx = blockIdx.x * blockDim.x + threadIdx.x;
    int total = C * Ncol;
    if (idx >= total) return;
    int e    = idx & 15;
    int lane = (idx >> 4) & 31;
    int frag = idx >> 9;
    int ntiles = Ncol >> 4;
    int nt    = frag % ntiles;
    int chunk = frag / ntiles;
    int hiK = lane >> 4, n16 = lane & 15;
    int k = chunk * 32 + wmma_kidx(e >> 1, e & 1, hiK);
    int n = nt * 16 + n16;
    int hh = n / d2, e2 = n % d2;
    out[idx] = (_Float16)w[((size_t)hh * C + k) * d2 + e2];
}

// plain weight (K, Ncol) fp32 -> B-frag f16
__global__ void pack_plain_sw_kernel(const float* __restrict__ w, _Float16* __restrict__ out,
                                     int K, int Ncol) {
    int idx = blockIdx.x * blockDim.x + threadIdx.x;
    int total = K * Ncol;
    if (idx >= total) return;
    int e    = idx & 15;
    int lane = (idx >> 4) & 31;
    int frag = idx >> 9;
    int ntiles = Ncol >> 4;
    int nt    = frag % ntiles;
    int chunk = frag / ntiles;
    int hiK = lane >> 4, n16 = lane & 15;
    int k = chunk * 32 + wmma_kidx(e >> 1, e & 1, hiK);
    int n = nt * 16 + n16;
    out[idx] = (_Float16)w[(size_t)k * Ncol + n];
}

// Q or K (B*N, h*2D) fp32 -> A-style frags per (b, head, 16-token tile, chunk, half)
// layout: ((((b*h+hh)*(N/16)+t)*KC+kc)*2+st)*512 + lane*16 + e ; zero-pad feat >= D
__global__ void swizzleQK_kernel(const float* __restrict__ x, _Float16* __restrict__ out,
                                 int Bn, int h, int N, int D, int KC) {
    int idx = blockIdx.x * blockDim.x + threadIdx.x;
    int total = Bn * h * (N >> 4) * KC * 2 * 512;
    if (idx >= total) return;
    int e    = idx & 15;
    int lane = (idx >> 4) & 31;
    int rest = idx >> 9;
    int st = rest & 1; rest >>= 1;
    int kc = rest % KC; rest /= KC;
    int t  = rest % (N >> 4); rest /= (N >> 4);
    int hh = rest % h;
    int b  = rest / h;
    int hiK = lane >> 4, n16 = lane & 15;
    int feat = kc * 32 + wmma_kidx(e >> 1, e & 1, hiK);
    int ld = h * 2 * D;
    float val = (feat < D)
        ? x[(size_t)(b * N + t * 16 + n16) * ld + hh * 2 * D + st * D + feat] : 0.0f;
    out[idx] = (_Float16)val;
}

// V (B*N, h*2D) fp32 -> B-style frags per (b, head, 32-key group, 16-col tile)
// layout: (((b*h+hh)*(N/32)+j)*NCV+cv)*512 + lane*16 + e
__global__ void swizzleV_kernel(const float* __restrict__ v, _Float16* __restrict__ out,
                                int Bn, int h, int N, int D, int NCV) {
    int idx = blockIdx.x * blockDim.x + threadIdx.x;
    int total = Bn * h * (N >> 5) * NCV * 512;
    if (idx >= total) return;
    int e    = idx & 15;
    int lane = (idx >> 4) & 31;
    int rest = idx >> 9;
    int cv = rest % NCV; rest /= NCV;
    int j  = rest % (N >> 5); rest /= (N >> 5);
    int hh = rest % h;
    int b  = rest / h;
    int hiK = lane >> 4, n16 = lane & 15;
    int kk = wmma_kidx(e >> 1, e & 1, hiK);
    int ld = h * 2 * D;
    out[idx] = (_Float16)v[(size_t)(b * N + j * 32 + kk) * ld + hh * 2 * D + cv * 16 + n16];
}

// (B,H,W,C) tokens -> NCHW output slice
__global__ void transpose_out_kernel(const float* __restrict__ t, float* __restrict__ o,
                                     int Bn, int C, int H, int W) {
    int idx = blockIdx.x * blockDim.x + threadIdx.x;
    int total = Bn * C * H * W;
    if (idx >= total) return;
    int x = idx % W;
    int y = (idx / W) % H;
    int c = (idx / (W * H)) % C;
    int b = idx / (W * H * C);
    o[idx] = t[(((size_t)b * H + y) * W + x) * C + c];
}

// ---------------------------------------------------------------------------
// WMMA GEMM on pre-swizzled fragments. 4 waves/block, one 16x16 tile per wave.
// flags: 1 = residual (out += old outF), 2 = GELU.
// ---------------------------------------------------------------------------
#define GFLAG_RES  1
#define GFLAG_GELU 2

__global__ __launch_bounds__(128) void gemm_wmma_kernel(
    const _Float16* __restrict__ A, const _Float16* __restrict__ Bw,
    const float* __restrict__ bias, float* outF, _Float16* outH,
    int M, int K, int Ncol, int flags) {
    const int lane = threadIdx.x & 31;
    const int wid  = threadIdx.x >> 5;
    const int hiK  = lane >> 4;
    const int n16  = lane & 15;
    const int bx = blockIdx.x;                 // col tile
    const int rt = blockIdx.y * 4 + wid;       // row tile
    const int chunks = K >> 5;
    const int ntiles = Ncol >> 4;
    (void)M;
    v8f acc = {};
    const _Float16* ap = A + ((size_t)rt * chunks) * 512 + lane * 16;
    for (int c = 0; c < chunks; ++c) {
        const _Float16* bp = Bw + ((size_t)c * ntiles + bx) * 512 + lane * 16;
        if (c + 1 < chunks) {
            __builtin_prefetch(ap + 512, 0, 3);
            __builtin_prefetch(bp + (size_t)ntiles * 512, 0, 3);
        }
        v16h a = *(const v16h*)ap;
        v16h b = *(const v16h*)bp;
        acc = __builtin_amdgcn_wmma_f32_16x16x32_f16(false, a, false, b,
                                                     (short)0, acc, false, false);
        ap += 512;
    }
    const int col = bx * 16 + n16;
#pragma unroll
    for (int vg = 0; vg < 8; ++vg) {
        int row = rt * 16 + vg + 8 * hiK;
        size_t idx = (size_t)row * Ncol + col;
        float v = acc[vg] + bias[col];
        if (flags & GFLAG_GELU) v = gelu_f(v);
        if (flags & GFLAG_RES)  v += outF[idx];
        if (outF) outF[idx] = v;
        if (outH) outH[idx] = (_Float16)v;
    }
}

// ---------------------------------------------------------------------------
// Differential flash attention on pre-swizzled Q/K/V fragments with a
// double-buffered TDM (tensor_load_to_lds) pipeline streaming K/V into LDS.
// One wave = one (b, head, 16-query tile); 32 keys per iteration; two
// independent streaming-softmax streams; o = (o1/l1 - lam*o2/l2)*(1-lam).
// ---------------------------------------------------------------------------
template <int D, int KC, int NCV>
__global__ __launch_bounds__(32) void attn_kernel(
    const _Float16* __restrict__ qfr, const _Float16* __restrict__ kfr,
    const _Float16* __restrict__ vfr, _Float16* __restrict__ o,
    const float* __restrict__ lam_p, int N, int h, int ld) {
    __shared__ _Float16 kT[2][KC * 2048];   // per-iter K frags: 2 subtiles x KC x 2 streams
    __shared__ _Float16 vT[2][NCV * 512];   // per-iter V frags
    __shared__ float    sS[16 * 32];
    __shared__ _Float16 sPf[512];           // P tile in A-fragment order
    __shared__ float sM1[16], sL1[16], sM2[16], sL2[16], sA[16];

    const int lane = threadIdx.x;
    const int hiK  = lane >> 4;
    const int n16  = lane & 15;
    const int qt = blockIdx.x;
    const int hh = blockIdx.y;
    const int b  = blockIdx.z;
    const int bh = b * h + hh;
    const int t16 = N >> 4, t32 = N >> 5;
    const float scale = rsqrtf((float)D);

    if (lane < 16) { sM1[lane] = -1e30f; sL1[lane] = 0.f; sM2[lane] = -1e30f; sL2[lane] = 0.f; }

    // Q fragments (resident across the key loop)
    v16h aq1[KC], aq2[KC];
    const _Float16* qbase = qfr + ((size_t)(bh * t16 + qt) * KC * 2) * 512 + lane * 16;
#pragma unroll
    for (int kc = 0; kc < KC; ++kc) {
        aq1[kc] = *(const v16h*)(qbase + (kc * 2 + 0) * 512);
        aq2[kc] = *(const v16h*)(qbase + (kc * 2 + 1) * 512);
    }

    v8f o1[NCV], o2[NCV];
    v8f zero = {};
#pragma unroll
    for (int c = 0; c < NCV; ++c) { o1[c] = zero; o2[c] = zero; }

    // TDM prologue: stage iteration 0 into buffer 0
    tdm_load_1d((unsigned)(size_t)&kT[0][0],
                kfr + ((size_t)(bh * t16) * KC * 2) * 512, KC * 2048);
    tdm_load_1d((unsigned)(size_t)&vT[0][0],
                vfr + ((size_t)(bh * t32) * NCV) * 512, NCV * 512);

    for (int j0 = 0; j0 < N; j0 += 32) {
        const int cur = (j0 >> 5) & 1;
        if (j0 + 32 < N) {
            // stage next iteration into the other buffer, then wait for current
            tdm_load_1d((unsigned)(size_t)&kT[cur ^ 1][0],
                        kfr + ((size_t)(bh * t16 + ((j0 + 32) >> 4)) * KC * 2) * 512,
                        KC * 2048);
            tdm_load_1d((unsigned)(size_t)&vT[cur ^ 1][0],
                        vfr + ((size_t)(bh * t32 + ((j0 + 32) >> 5)) * NCV) * 512,
                        NCV * 512);
            __builtin_amdgcn_s_wait_tensorcnt(2);
        } else {
            __builtin_amdgcn_s_wait_tensorcnt(0);
        }
        const _Float16* kTc = &kT[cur][0];
        const _Float16* vTc = &vT[cur][0];
#pragma unroll
        for (int st = 0; st < 2; ++st) {
            // --- scores for two 16-key sub-tiles (K from LDS) ---
#pragma unroll
            for (int sub = 0; sub < 2; ++sub) {
                v8f s = {};
#pragma unroll
                for (int kc = 0; kc < KC; ++kc) {
                    v16h bk = *(const v16h*)(kTc + ((sub * KC + kc) * 2 + st) * 512 + lane * 16);
                    s = __builtin_amdgcn_wmma_f32_16x16x32_f16(
                            false, (st ? aq2[kc] : aq1[kc]), false, bk,
                            (short)0, s, false, false);
                }
#pragma unroll
                for (int vg = 0; vg < 8; ++vg)
                    sS[(vg + 8 * hiK) * 32 + sub * 16 + n16] = s[vg] * scale;
            }
            // --- streaming softmax row update; scatter P in A-frag order ---
            float* sM = st ? sM2 : sM1;
            float* sL = st ? sL2 : sL1;
            if (lane < 16) {
                float tm = -1e30f;
                for (int c = 0; c < 32; ++c) tm = fmaxf(tm, sS[lane * 32 + c]);
                float nm = fmaxf(sM[lane], tm);
                float al = __expf(sM[lane] - nm);
                float sum = 0.f;
                for (int c = 0; c < 32; ++c) {
                    float p = __expf(sS[lane * 32 + c] - nm);
                    int hi   = (c >> 3) & 1;
                    int low3 = c & 7;
                    int vg   = (c >> 4) * 4 + (low3 >> 1);
                    sPf[(lane + 16 * hi) * 16 + vg * 2 + (low3 & 1)] = (_Float16)p;
                    sum += p;
                }
                sL[lane] = al * sL[lane] + sum;
                sM[lane] = nm;
                sA[lane] = al;
            }
            // --- rescale O and accumulate P x V (P + V from LDS) ---
            v16h pa = *(const v16h*)(sPf + lane * 16);
#pragma unroll
            for (int cv = 0; cv < NCV; ++cv) {
                v8f oc = st ? o2[cv] : o1[cv];
#pragma unroll
                for (int vg = 0; vg < 8; ++vg) oc[vg] *= sA[vg + 8 * hiK];
                v16h bv = *(const v16h*)(vTc + cv * 512 + lane * 16);
                oc = __builtin_amdgcn_wmma_f32_16x16x32_f16(false, pa, false, bv,
                                                            (short)0, oc, false, false);
                if (st) o2[cv] = oc; else o1[cv] = oc;
            }
        }
    }
    // --- finalize ---
    const float lam = *lam_p;
    const float om  = 1.f - lam;
#pragma unroll
    for (int cv = 0; cv < NCV; ++cv)
#pragma unroll
        for (int vg = 0; vg < 8; ++vg) {
            int row = vg + 8 * hiK;
            float val = (o1[cv][vg] / sL1[row] - lam * o2[cv][vg] / sL2[row]) * om;
            o[(size_t)(b * N + qt * 16 + row) * ld + hh * 2 * D + cv * 16 + n16] = (_Float16)val;
        }
}

// ---------------------------------------------------------------------------
// Host orchestration
// ---------------------------------------------------------------------------
extern "C" void kernel_launch(void* const* d_in, const int* in_sizes, int n_in,
                              void* d_out, int out_size, void* d_ws, size_t ws_size,
                              hipStream_t stream) {
    (void)in_sizes; (void)n_in; (void)out_size; (void)ws_size;
    const int B = 4;
    const int dims[4]   = {32, 64, 160, 256};
    const int heads[4]  = {1, 2, 4, 8};
    const int depths[4] = {3, 4, 6, 3};
    const int ksz[4]    = {7, 3, 3, 3};
    const int strd[4]   = {4, 2, 2, 2};
    const int Hs[4]     = {64, 32, 16, 8};

    char* base = (char*)d_ws;
    size_t off = 0;
    auto walloc = [&](size_t bytes) -> void* {
        off = (off + 255) & ~(size_t)255;
        void* p = base + off;
        off += bytes;
        return p;
    };
    float*    t0     = (float*)walloc((size_t)524288 * 4);
    float*    t1     = (float*)walloc((size_t)524288 * 4);
    float*    qbuf   = (float*)walloc((size_t)1048576 * 4);
    float*    kbuf   = (float*)walloc((size_t)1048576 * 4);
    float*    vbuf   = (float*)walloc((size_t)1048576 * 4);
    _Float16* act_h  = (_Float16*)walloc((size_t)524288 * 2);
    _Float16* act_sw = (_Float16*)walloc((size_t)524288 * 2);
    _Float16* o_h    = (_Float16*)walloc((size_t)1048576 * 2);
    _Float16* o_sw   = (_Float16*)walloc((size_t)1048576 * 2);
    _Float16* hid_h  = (_Float16*)walloc((size_t)2097152 * 2);
    _Float16* hid_sw = (_Float16*)walloc((size_t)2097152 * 2);
    _Float16* qfr    = (_Float16*)walloc((size_t)1048576 * 2);
    _Float16* kfr    = (_Float16*)walloc((size_t)1048576 * 2);
    _Float16* vfr    = (_Float16*)walloc((size_t)1048576 * 2);
    _Float16* wq_h   = (_Float16*)walloc((size_t)131072 * 2);
    _Float16* wk_h   = (_Float16*)walloc((size_t)131072 * 2);
    _Float16* wv_h   = (_Float16*)walloc((size_t)131072 * 2);
    _Float16* wo_h   = (_Float16*)walloc((size_t)131072 * 2);
    _Float16* w1_h   = (_Float16*)walloc((size_t)262144 * 2);
    _Float16* w2_h   = (_Float16*)walloc((size_t)262144 * 2);

    int cur = 0;
    const float* xin = (const float*)d_in[cur++];
    float* outp = (float*)d_out;

    hipMemcpyAsync(outp, xin, (size_t)786432 * sizeof(float),
                   hipMemcpyDeviceToDevice, stream);
    size_t outOff = 786432;

    const float* prevFeat = xin;
    int prevNCHW = 1, Cin = 3, Hin = 256;
    float* tbuf[2] = {t0, t1};

    for (int i = 0; i < 4; ++i) {
        const float* conv_w = (const float*)d_in[cur++];
        const float* conv_b = (const float*)d_in[cur++];
        const float* ln_w   = (const float*)d_in[cur++];
        const float* ln_b   = (const float*)d_in[cur++];
        const float* lam    = (const float*)d_in[cur++];

        float* t = tbuf[i & 1];
        const int C = dims[i], h = heads[i], dep = depths[i];
        const int Ho = Hs[i], N = Ho * Ho, M = B * N;
        const int d = C / h, d2 = 2 * d, h2d = h * d2, C4 = 4 * C;
        const int KC = (d > 32) ? 2 : 1;
        const int NCV = d2 / 16;

        {
            int total = B * Ho * Ho * C;
            conv_kernel<<<(total + 255) / 256, 256, 0, stream>>>(
                prevFeat, conv_w, conv_b, t, B, Cin, Hin, Hin, C,
                Ho, Ho, ksz[i], strd[i], ksz[i] / 2, prevNCHW);
        }
        ln_rms_kernel<<<(M + 7) / 8, 256, 0, stream>>>(t, ln_w, ln_b, M, C);

        for (int L = 0; L < dep; ++L) {
            const float* Wq = (const float*)d_in[cur++];
            const float* bq = (const float*)d_in[cur++];
            const float* Wk = (const float*)d_in[cur++];
            const float* bk = (const float*)d_in[cur++];
            const float* Wv = (const float*)d_in[cur++];
            const float* bv = (const float*)d_in[cur++];
            const float* Wo = (const float*)d_in[cur++];
            const float* bo = (const float*)d_in[cur++];
            const float* W1 = (const float*)d_in[cur++];
            const float* b1 = (const float*)d_in[cur++];
            const float* W2 = (const float*)d_in[cur++];
            const float* b2 = (const float*)d_in[cur++];

            int nqkv = C * h2d;
            pack_qkv_sw_kernel<<<(nqkv + 255) / 256, 256, 0, stream>>>(Wq, wq_h, h, C, d2);
            pack_qkv_sw_kernel<<<(nqkv + 255) / 256, 256, 0, stream>>>(Wk, wk_h, h, C, d2);
            pack_qkv_sw_kernel<<<(nqkv + 255) / 256, 256, 0, stream>>>(Wv, wv_h, h, C, d2);
            pack_plain_sw_kernel<<<(h2d * C + 255) / 256, 256, 0, stream>>>(Wo, wo_h, h2d, C);
            pack_plain_sw_kernel<<<(C * C4 + 255) / 256, 256, 0, stream>>>(W1, w1_h, C, C4);
            pack_plain_sw_kernel<<<(C4 * C + 255) / 256, 256, 0, stream>>>(W2, w2_h, C4, C);

            // xi = rmsnorm(t) -> f16, then A-fragment swizzle
            rms_half_kernel<<<(M + 7) / 8, 256, 0, stream>>>(t, act_h, M, C);
            swizzleA_kernel<<<(M * C + 255) / 256, 256, 0, stream>>>(act_h, act_sw, M, C);

            // QKV projections (fp32 out)
            gemm_wmma_kernel<<<dim3(h2d / 16, M / 64), 128, 0, stream>>>(
                act_sw, wq_h, bq, qbuf, nullptr, M, C, h2d, 0);
            gemm_wmma_kernel<<<dim3(h2d / 16, M / 64), 128, 0, stream>>>(
                act_sw, wk_h, bk, kbuf, nullptr, M, C, h2d, 0);
            gemm_wmma_kernel<<<dim3(h2d / 16, M / 64), 128, 0, stream>>>(
                act_sw, wv_h, bv, vbuf, nullptr, M, C, h2d, 0);

            // fragment-swizzle Q/K/V for the attention inner loop
            {
                int tq = B * h * (N >> 4) * KC * 2 * 512;
                swizzleQK_kernel<<<(tq + 255) / 256, 256, 0, stream>>>(qbuf, qfr, B, h, N, d, KC);
                swizzleQK_kernel<<<(tq + 255) / 256, 256, 0, stream>>>(kbuf, kfr, B, h, N, d, KC);
                int tv = B * h * (N >> 5) * NCV * 512;
                swizzleV_kernel<<<(tv + 255) / 256, 256, 0, stream>>>(vbuf, vfr, B, h, N, d, NCV);
            }

            dim3 ag(N / 16, h, B);
            if (d == 40)
                attn_kernel<40, 2, 5><<<ag, 32, 0, stream>>>(qfr, kfr, vfr, o_h, lam, N, h, h2d);
            else
                attn_kernel<32, 1, 4><<<ag, 32, 0, stream>>>(qfr, kfr, vfr, o_h, lam, N, h, h2d);

            // t += o @ Wo + bo
            swizzleA_kernel<<<(M * h2d + 255) / 256, 256, 0, stream>>>(o_h, o_sw, M, h2d);
            gemm_wmma_kernel<<<dim3(C / 16, M / 64), 128, 0, stream>>>(
                o_sw, wo_h, bo, t, nullptr, M, h2d, C, GFLAG_RES);

            // MLP
            rms_half_kernel<<<(M + 7) / 8, 256, 0, stream>>>(t, act_h, M, C);
            swizzleA_kernel<<<(M * C + 255) / 256, 256, 0, stream>>>(act_h, act_sw, M, C);
            gemm_wmma_kernel<<<dim3(C4 / 16, M / 64), 128, 0, stream>>>(
                act_sw, w1_h, b1, nullptr, hid_h, M, C, C4, GFLAG_GELU);
            swizzleA_kernel<<<(M * C4 + 255) / 256, 256, 0, stream>>>(hid_h, hid_sw, M, C4);
            gemm_wmma_kernel<<<dim3(C / 16, M / 64), 128, 0, stream>>>(
                hid_sw, w2_h, b2, t, nullptr, M, C4, C, GFLAG_RES);
        }

        {
            int total = B * C * Ho * Ho;
            transpose_out_kernel<<<(total + 255) / 256, 256, 0, stream>>>(
                t, outp + outOff, B, C, Ho, Ho);
            outOff += (size_t)total;
        }
        prevFeat = t;
        prevNCHW = 0;
        Cin = C;
        Hin = Ho;
    }
}
